// SelfAttention_21122649161760
// MI455X (gfx1250) — matrix operand order repaired and checked by reading, not compiled, not run on hardware
//
#include <hip/hip_runtime.h>
#include <math.h>

typedef __attribute__((ext_vector_type(16))) _Float16 v16h;
typedef __attribute__((ext_vector_type(8)))  _Float16 v8h;
typedef __attribute__((ext_vector_type(8)))  float    v8f;

#define B_    2
#define S_    2048
#define H_    16
#define D_    128
#define NEGV  (-10000.0f)
#define QTILE 256
#define KTILE 32
#define NWAVE 8
#define MT    2   // M-tiles (16 query rows each) per wave

// Process one 32-key block for M-tiles [MTB, MT) of this wave.
// MTB=0: both tiles live; MTB=1: only the upper tile is below the diagonal.
template<int MTB>
__device__ __forceinline__ void fa_block(
    const v16h (&qA)[MT][4], const v16h& ones, v8f (&o)[MT][8],
    float (&m_i)[MT][8], float (&l_i)[MT][8],
    const _Float16* __restrict__ sKb, const _Float16* __restrict__ sVtb,
    _Float16* __restrict__ myP,
    int kb, int qfirstW, int qi_baseW, int half, int l16, int off)
{
    // ---------------- S = Q @ K^T : share each K B-fragment across M-tiles ----------------
    v8f s0[MT], s1[MT];
#pragma unroll
    for (int mt = MTB; mt < MT; ++mt)
#pragma unroll
        for (int r = 0; r < 8; ++r) { s0[mt][r] = 0.0f; s1[mt][r] = 0.0f; }

#pragma unroll
    for (int c = 0; c < 4; ++c) {
        // B frag: element e <-> K = half*16 + e, column n = l16 (+16 for s1)
        const v16h bk0 = *(const v16h*)&sKb[l16 * D_        + c * 32 + half * 16];
        const v16h bk1 = *(const v16h*)&sKb[(16 + l16) * D_ + c * 32 + half * 16];
#pragma unroll
        for (int mt = MTB; mt < MT; ++mt) {
            s0[mt] = __builtin_amdgcn_wmma_f32_16x16x32_f16(false, qA[mt][c], false, bk0, (short)0, s0[mt], false, false);
            s1[mt] = __builtin_amdgcn_wmma_f32_16x16x32_f16(false, qA[mt][c], false, bk1, (short)0, s1[mt], false, false);
        }
    }

    // ---------------- causal mask + online softmax per M-tile ----------------
    v16h pA[MT];
    float alpha[MT][8];
#pragma unroll
    for (int mt = MTB; mt < MT; ++mt) {
        const bool needMask = (kb + KTILE - 1) > (qfirstW + mt * 128); // wave-uniform
        const int kj0 = kb + l16;
        const int kj1 = kb + 16 + l16;
        float p0[8], p1[8];
#pragma unroll
        for (int r = 0; r < 8; ++r) {
            float x0 = s0[mt][r];
            float x1 = s1[mt][r];
            if (needMask) {
                const int qi = qi_baseW + mt * 128 + r;
                x0 += (kj0 > qi) ? NEGV : 0.0f;
                x1 += (kj1 > qi) ? NEGV : 0.0f;
            }
            float rm = fmaxf(x0, x1);
            rm = fmaxf(rm, __shfl_xor(rm, 1, 32));
            rm = fmaxf(rm, __shfl_xor(rm, 2, 32));
            rm = fmaxf(rm, __shfl_xor(rm, 4, 32));
            rm = fmaxf(rm, __shfl_xor(rm, 8, 32));
            const float mnew = fmaxf(m_i[mt][r], rm);
            alpha[mt][r] = __expf(m_i[mt][r] - mnew);
            p0[r]        = __expf(x0 - mnew);
            p1[r]        = __expf(x1 - mnew);
            m_i[mt][r]   = mnew;
        }
#pragma unroll
        for (int f = 0; f < 8; ++f)
#pragma unroll
            for (int r = 0; r < 8; ++r) o[mt][f][r] *= alpha[mt][r];

        // P (f32 C layout) -> f16 A layout via per-wave-per-tile LDS scratch
        _Float16* P = myP + mt * (16 * KTILE);
#pragma unroll
        for (int r = 0; r < 8; ++r) {
            const int m = r + half * 8;
            P[m * KTILE + l16]      = (_Float16)p0[r];
            P[m * KTILE + 16 + l16] = (_Float16)p1[r];
        }
        // same-wave DS ops are in-order: safe without a barrier
        const v8h lo = *(const v8h*)&P[l16 * KTILE + 2 * off];
        const v8h hi = *(const v8h*)&P[l16 * KTILE + 2 * off + 8];
#pragma unroll
        for (int e = 0; e < 8; ++e) { pA[mt][e] = lo[e]; pA[mt][8 + e] = hi[e]; }

        // rowsum(P) via WMMA with a ones B-matrix: every column of the result
        // holds the row sum, already in the (m = r + 8*half) per-lane layout.
        v8f zsum;
#pragma unroll
        for (int r = 0; r < 8; ++r) zsum[r] = 0.0f;
        zsum = __builtin_amdgcn_wmma_f32_16x16x32_f16(false, pA[mt], false, ones, (short)0, zsum, false, false);
#pragma unroll
        for (int r = 0; r < 8; ++r)
            l_i[mt][r] = l_i[mt][r] * alpha[mt][r] + zsum[r];
    }

    // ---------------- O += P @ V : share each V B-fragment across M-tiles ----------------
#pragma unroll
    for (int f = 0; f < 8; ++f) {
        // B frag: element e <-> key = half*16 + e, column dim = f*16 + l16
        const v16h bv = *(const v16h*)&sVtb[(f * 16 + l16) * KTILE + half * 16];
#pragma unroll
        for (int mt = MTB; mt < MT; ++mt)
            o[mt][f] = __builtin_amdgcn_wmma_f32_16x16x32_f16(false, pA[mt], false, bv, (short)0, o[mt][f], false, false);
    }
}

__global__ __launch_bounds__(256) void fa_fwd_gfx1250(const float* __restrict__ qkv,
                                                      float* __restrict__ out)
{
    // double-buffered K/V tiles + per-wave-per-tile P scratch: 16+16+16 = 48 KB LDS
    __shared__ __align__(32) _Float16 sK [2][KTILE * D_];   // [key][dim] row-major f16
    __shared__ __align__(32) _Float16 sVt[2][D_ * KTILE];   // [dim][key] transposed f16
    __shared__ __align__(32) _Float16 sP [NWAVE * MT * 16 * KTILE];

    const int tid  = threadIdx.x;
    const int wave = tid >> 5;
    const int lane = tid & 31;
    const int half = lane >> 4;
    const int l16  = lane & 15;
    const int off  = half * 8;

    const int bh = blockIdx.y;
    const int b  = bh / H_;
    const int h  = bh % H_;
    const int q0 = blockIdx.x * QTILE;

    const float scale = 0.08838834764831845f; // 1/sqrt(128)

    // ---------------- Q fragments (A layout) for both M-tiles, pre-scaled ----------------
    v16h qA[MT][4];
#pragma unroll
    for (int mt = 0; mt < MT; ++mt) {
        const int qrow = q0 + mt * 128 + wave * 16 + l16;
        const float* qptr = qkv + ((((size_t)b * S_ + qrow) * 3 + 0) * H_ + h) * D_;
#pragma unroll
        for (int c = 0; c < 4; ++c) {
            const float* p0 = qptr + c * 32 + 2 * off;
            const float* p1 = qptr + c * 32 + 2 * off + 8;
#pragma unroll
            for (int e = 0; e < 8; ++e) {
                qA[mt][c][e]     = (_Float16)(p0[e] * scale);
                qA[mt][c][8 + e] = (_Float16)(p1[e] * scale);
            }
        }
    }

    v16h ones;
#pragma unroll
    for (int e = 0; e < 16; ++e) ones[e] = (_Float16)1.0f;

    // ---------------- accumulators ----------------
    v8f o[MT][8];
    float m_i[MT][8], l_i[MT][8];
#pragma unroll
    for (int mt = 0; mt < MT; ++mt) {
#pragma unroll
        for (int f = 0; f < 8; ++f)
#pragma unroll
            for (int r = 0; r < 8; ++r) o[mt][f][r] = 0.0f;
#pragma unroll
        for (int r = 0; r < 8; ++r) { m_i[mt][r] = -3.0e38f; l_i[mt][r] = 0.0f; }
    }

    const int qfirstW  = q0 + wave * 16;            // lowest row of M-tile 0
    const int qi_baseW = qfirstW + half * 8;        // C-frag row r -> query qi_baseW + mt*128 + r
    const int qlast0   = qfirstW + 15;
    const int qlast1   = qlast0 + 128;
    const int nkb      = (q0 + QTILE) / KTILE;

    // cooperative tile staging (256 threads, 32 keys x 128 dims)
    const int ldKey = tid >> 3;
    const int ldSeg = (tid & 7) * 16;
    const size_t rowStride = (size_t)3 * H_ * D_;
    const float* kBase0 = qkv + (((size_t)b * S_ * 3 + 1) * H_ + h) * D_ + ldSeg;
    const float* vBase0 = qkv + (((size_t)b * S_ * 3 + 2) * H_ + h) * D_ + ldSeg;

    float4 kr[4], vr[4];
    auto loadTile = [&](int kb) {
        const float4* kp = (const float4*)(kBase0 + (size_t)(kb + ldKey) * rowStride);
        const float4* vp = (const float4*)(vBase0 + (size_t)(kb + ldKey) * rowStride);
#pragma unroll
        for (int i = 0; i < 4; ++i) { kr[i] = kp[i]; vr[i] = vp[i]; }
    };
    auto storeTile = [&](int buf) {
        const float* kf = (const float*)kr;
        const float* vf = (const float*)vr;
        v8h hk0, hk1;
#pragma unroll
        for (int e = 0; e < 8; ++e) { hk0[e] = (_Float16)kf[e]; hk1[e] = (_Float16)kf[8 + e]; }
        *(v8h*)&sK[buf][ldKey * D_ + ldSeg]     = hk0;
        *(v8h*)&sK[buf][ldKey * D_ + ldSeg + 8] = hk1;
#pragma unroll
        for (int e = 0; e < 16; ++e)
            sVt[buf][(ldSeg + e) * KTILE + ldKey] = (_Float16)vf[e];
    };

    loadTile(0);
    storeTile(0);
    __syncthreads();

    _Float16* myP = &sP[wave * MT * 16 * KTILE];

    for (int kbi = 0; kbi < nkb; ++kbi) {
        const int  cur  = kbi & 1;
        const int  kb   = kbi * KTILE;
        const bool more = (kbi + 1) < nkb;

        if (more) loadTile(kb + KTILE);                       // overlap with WMMA below
        if (kbi + 2 < nkb) {                                  // global_prefetch_b8, tile i+2
            __builtin_prefetch(kBase0 + (size_t)(kb + 2 * KTILE + ldKey) * rowStride, 0, 1);
            __builtin_prefetch(vBase0 + (size_t)(kb + 2 * KTILE + ldKey) * rowStride, 0, 1);
        }

        if (kb <= qlast0) {        // both M-tiles live
            fa_block<0>(qA, ones, o, m_i, l_i, sK[cur], sVt[cur], myP,
                        kb, qfirstW, qi_baseW, half, l16, off);
        } else if (kb <= qlast1) { // only upper M-tile live
            fa_block<1>(qA, ones, o, m_i, l_i, sK[cur], sVt[cur], myP,
                        kb, qfirstW, qi_baseW, half, l16, off);
        }

        if (more) storeTile(1 - cur);
        __syncthreads();
    }

    // ---------------- epilogue: normalize, store f32 (bthd) ----------------
#pragma unroll
    for (int mt = 0; mt < MT; ++mt) {
        float inv[8];
#pragma unroll
        for (int r = 0; r < 8; ++r) inv[r] = 1.0f / l_i[mt][r];
#pragma unroll
        for (int f = 0; f < 8; ++f) {
            const int d = f * 16 + l16;
#pragma unroll
            for (int r = 0; r < 8; ++r) {
                const int row = qi_baseW + mt * 128 + r;
                out[(((size_t)b * S_ + row) * H_ + h) * D_ + d] = o[mt][f][r] * inv[r];
            }
        }
    }
}

extern "C" void kernel_launch(void* const* d_in, const int* in_sizes, int n_in,
                              void* d_out, int out_size, void* d_ws, size_t ws_size,
                              hipStream_t stream) {
    const float* qkv = (const float*)d_in[0];
    float* out = (float*)d_out;
    dim3 grid(S_ / QTILE, B_ * H_);
    fa_fwd_gfx1250<<<grid, 256, 0, stream>>>(qkv, out);
}
